// Decoder_My_54047868453114
// MI455X (gfx1250) — compile-verified, matrix operand order checked by reading, not run on hardware
//
#include <hip/hip_runtime.h>
#include <hip/hip_bf16.h>

// ---------------- problem constants ----------------
#define PB   256   // batch
#define PN   512   // nodes
#define PD   512   // embed dim
#define PH   512   // hidden
#define PT   128   // decode steps
#define PHEADS 8
#define PKEY 64
#define K2PAD 544  // 515 padded to multiple of 32

typedef __attribute__((ext_vector_type(16))) __bf16 v16bf;
typedef __attribute__((ext_vector_type(8)))  float  v8f;
typedef __attribute__((ext_vector_type(8)))  unsigned int v8u;

// ---------------- scalar helpers ----------------
__device__ __forceinline__ unsigned short f2bf(float f) {
  unsigned int u = __builtin_bit_cast(unsigned int, f);
  unsigned int r = u + 0x7FFFu + ((u >> 16) & 1u);
  return (unsigned short)(r >> 16);
}
__device__ __forceinline__ v8f v8f_zero() {
  v8f z;
#pragma unroll
  for (int i = 0; i < 8; ++i) z[i] = 0.0f;
  return z;
}
__device__ __forceinline__ v8f wmma_bf16(v16bf a, v16bf b, v8f c) {
  return __builtin_amdgcn_wmma_f32_16x16x32_bf16(
      false, a, false, b, (short)0, c, false, false);
}
// Fragment = 16 contiguous bf16 at p (already lane-offset).
__device__ __forceinline__ v16bf load_frag_at(const unsigned short* p) {
  return __builtin_bit_cast(v16bf, *(const v8u*)p);
}
__device__ __forceinline__ v16bf load_frag(const unsigned short* p) {
  return load_frag_at(p + (threadIdx.x & 31) * 16);
}
// Packed-tile position for element (m in 0..15, k in 0..31) of a 16x32 frag tile.
__device__ __forceinline__ int frag_idx(int m, int k) {
  int lane = (m & 15) + 16 * ((k >> 3) & 1);
  int pos = (k & 7) + 8 * ((k >> 4) & 1);
  return lane * 16 + pos;
}
// gfx1250 transcendental tanh (V_TANH_F32); v_nop covers TRANS->use hazard.
__device__ __forceinline__ float fast_tanh(float x) {
  float r;
  asm volatile("v_tanh_f32 %0, %1\n\tv_nop" : "=v"(r) : "v"(x));
  return r;
}

// ---------------- packing kernels ----------------
__global__ void pack_a_kernel(const float* __restrict__ src,
                              unsigned short* __restrict__ dst, int M, int K) {
  int KT = K >> 5;
  size_t chunks = (size_t)M * (K >> 3);
  size_t i = (size_t)blockIdx.x * blockDim.x + threadIdx.x;
  size_t stride = (size_t)gridDim.x * blockDim.x;
  for (; i < chunks; i += stride) {
    int r = (int)(i / (K >> 3));
    int c0 = ((int)(i % (K >> 3))) << 3;
    int mt = r >> 4, kt = c0 >> 5;
    int base = frag_idx(r & 15, c0 & 31);
    unsigned short tmp[8];
#pragma unroll
    for (int e = 0; e < 8; ++e) tmp[e] = f2bf(src[(size_t)r * K + c0 + e]);
    *(uint4*)(dst + ((size_t)mt * KT + kt) * 512 + base) = *(const uint4*)tmp;
  }
}

__global__ void pack_b_kernel(const float* __restrict__ src,
                              unsigned short* __restrict__ dst,
                              int K, int N, int Kvalid) {
  int KT = K >> 5;
  size_t chunks = (size_t)N * (K >> 3);
  size_t i = (size_t)blockIdx.x * blockDim.x + threadIdx.x;
  size_t stride = (size_t)gridDim.x * blockDim.x;
  for (; i < chunks; i += stride) {
    int n = (int)(i / (K >> 3));
    int k0 = ((int)(i % (K >> 3))) << 3;
    int nt = n >> 4, kt = k0 >> 5;
    int base = frag_idx(n & 15, k0 & 31);
    unsigned short tmp[8];
#pragma unroll
    for (int e = 0; e < 8; ++e) {
      int k = k0 + e;
      tmp[e] = (k < Kvalid) ? f2bf(src[(size_t)k * N + n]) : (unsigned short)0;
    }
    *(uint4*)(dst + ((size_t)nt * KT + kt) * 512 + base) = *(const uint4*)tmp;
  }
}

__global__ void next_nodes_kernel(const float* __restrict__ emb,
                                  const int* __restrict__ cur,
                                  const float* __restrict__ ucap,
                                  const float* __restrict__ ubat,
                                  const float* __restrict__ ctime,
                                  unsigned short* __restrict__ dst) {
  const int K = K2PAD, KT = K2PAD >> 5;
  size_t chunks = (size_t)PB * PT * (K >> 3);
  size_t i = (size_t)blockIdx.x * blockDim.x + threadIdx.x;
  size_t stride = (size_t)gridDim.x * blockDim.x;
  for (; i < chunks; i += stride) {
    int row = (int)(i / (K >> 3));
    int c0 = ((int)(i % (K >> 3))) << 3;
    int b = row >> 7;
    int node = cur[row];
    unsigned short tmp[8];
#pragma unroll
    for (int e = 0; e < 8; ++e) {
      int c = c0 + e;
      float v;
      if (c < PD)            v = emb[((size_t)b * PN + node) * PD + c];
      else if (c == PD)      v = 1.0f - ucap[row];
      else if (c == PD + 1)  v = 1.0f - ubat[row];
      else if (c == PD + 2)  v = ctime[row];
      else                   v = 0.0f;
      tmp[e] = f2bf(v);
    }
    int mt = row >> 4, kt = c0 >> 5;
    int base = frag_idx(row & 15, c0 & 31);
    *(uint4*)(dst + ((size_t)mt * KT + kt) * 512 + base) = *(const uint4*)tmp;
  }
}

__global__ void mean_embed_kernel(const float* __restrict__ emb,
                                  float* __restrict__ meanE) {
  int b = blockIdx.x;
  for (int d = threadIdx.x; d < PD; d += blockDim.x) {
    float s = 0.0f;
    for (int n = 0; n < PN; ++n) s += emb[((size_t)b * PN + n) * PD + d];
    meanE[(size_t)b * PD + d] = s * (1.0f / PN);
  }
}

__global__ void fixed_context_kernel(const float* __restrict__ meanE,
                                     const float* __restrict__ Wc,
                                     float* __restrict__ fixed) {
  int b = blockIdx.x;
  for (int h = threadIdx.x; h < PH; h += blockDim.x) {
    float s = 0.0f;
    for (int d = 0; d < PD; ++d)
      s += meanE[(size_t)b * PD + d] * Wc[(size_t)d * PH + h];
    fixed[(size_t)b * PH + h] = s;
  }
}

__global__ void query_epi_kernel(const float* __restrict__ q,
                                 const float* __restrict__ fixed,
                                 unsigned short* __restrict__ dst) {
  size_t chunks = (size_t)PB * PT * (PH >> 3);
  size_t i = (size_t)blockIdx.x * blockDim.x + threadIdx.x;
  size_t stride = (size_t)gridDim.x * blockDim.x;
  for (; i < chunks; i += stride) {
    int row = (int)(i >> 6);
    int c0 = ((int)(i & 63)) << 3;
    int b = row >> 7;
    unsigned short tmp[8];
#pragma unroll
    for (int e = 0; e < 8; ++e)
      tmp[e] = f2bf(q[(size_t)row * PH + c0 + e] +
                    fixed[(size_t)b * PH + c0 + e]);
    int mt = row >> 4, kt = c0 >> 5;
    int base = frag_idx(row & 15, c0 & 31);
    *(uint4*)(dst + ((size_t)mt * 16 + kt) * 512 + base) = *(const uint4*)tmp;
  }
}

// ---------------- LDS-free packed WMMA GEMM (compile-time KT / NCOLS) --------
// Block 256 = 8 waves (4 along M x 2 along N); wave tile 32x64 -> block 128x128.
// KT/NCOLS constexpr => all fragment/store offsets fold into immediates.
template <int KT, int NCOLS, int STORE_BF16>
__global__ __launch_bounds__(256)
void gemm_packed_kernel(const unsigned short* __restrict__ A,
                        const unsigned short* __restrict__ B,
                        void* __restrict__ Cout) {
  int tid = threadIdx.x;
  int lane = tid & 31;
  int wid = tid >> 5;
  int wm = wid & 3, wn = wid >> 2;
  int mt0 = blockIdx.y * 8 + wm * 2;
  int ct0 = blockIdx.x * 8 + wn * 4;

  v8f acc[2][4];
#pragma unroll
  for (int u = 0; u < 2; ++u)
#pragma unroll
    for (int j = 0; j < 4; ++j) acc[u][j] = v8f_zero();

  const unsigned short* a0p = A + (size_t)mt0 * KT * 512 + lane * 16;
  const unsigned short* a1p = a0p + (size_t)KT * 512;
  const unsigned short* bp  = B + (size_t)ct0 * KT * 512 + lane * 16;
#pragma unroll 4
  for (int kt = 0; kt < KT; ++kt) {
    v16bf af0 = load_frag_at(a0p + kt * 512);
    v16bf af1 = load_frag_at(a1p + kt * 512);
#pragma unroll
    for (int j = 0; j < 4; ++j) {
      v16bf bf_ = load_frag_at(bp + (j * KT + kt) * 512);
      acc[0][j] = wmma_bf16(af0, bf_, acc[0][j]);
      acc[1][j] = wmma_bf16(af1, bf_, acc[1][j]);
    }
  }

  int n = lane & 15;
  int mbase = (lane >> 4) << 3;
  size_t cbase = (size_t)(mt0 * 16 + mbase) * NCOLS + ct0 * 16 + n;
  if (STORE_BF16) {
    unsigned short* C = (unsigned short*)Cout + cbase;
#pragma unroll
    for (int u = 0; u < 2; ++u)
#pragma unroll
      for (int j = 0; j < 4; ++j)
#pragma unroll
        for (int i = 0; i < 8; ++i)
          C[(size_t)(u * 16 + i) * NCOLS + j * 16] = f2bf(acc[u][j][i]);
  } else {
    float* C = (float*)Cout + cbase;
#pragma unroll
    for (int u = 0; u < 2; ++u)
#pragma unroll
      for (int j = 0; j < 4; ++j)
#pragma unroll
        for (int i = 0; i < 8; ++i)
          C[(size_t)(u * 16 + i) * NCOLS + j * 16] = acc[u][j][i];
  }
}

// Stage a 16-row x cols (mult of 32) row-major global tile into LDS frag tiles
// using CDNA5 async global->LDS DMA (contiguous 16B chunks, ASYNCcnt-tracked).
__device__ __forceinline__ void stage_t16_async(unsigned short* __restrict__ dst,
                                                const unsigned short* __restrict__ src,
                                                size_t ld, int cols, int lane) {
  int cpr = cols >> 3;  // 16B chunks per row
  int chunks = 16 * cpr;
  for (int i = lane; i < chunks; i += 32) {
    int r = i / cpr;
    int c0 = (i % cpr) << 3;
    const unsigned short* gp = src + (size_t)r * ld + c0;
    unsigned short* lp = dst + ((c0 >> 5) * 512) + frag_idx(r, c0 & 31);
    unsigned int ldsoff = (unsigned int)(unsigned long long)lp;
    asm volatile("global_load_async_to_lds_b128 %0, %1, off"
                 :
                 : "v"(ldsoff), "v"((unsigned long long)gp)
                 : "memory");
  }
  asm volatile("s_wait_asynccnt 0" ::: "memory");
}

// ---------------- fused MHA glimpse ----------------
// One wave per (b, head, 16-row T tile). grid = B*HEADS*(T/16) = 16384.
__global__ __launch_bounds__(32)
void attention_kernel(const unsigned short* __restrict__ qP,
                      const unsigned short* __restrict__ kvl,
                      const unsigned char* __restrict__ mask,
                      unsigned short* __restrict__ headsP) {
  __shared__ __align__(32) float          sScore[16 * 512];  // 32 KB
  __shared__ __align__(32) unsigned short sProbP[16 * 512];  // 16 KB
  __shared__ __align__(32) unsigned short sK[2 * 512];
  __shared__ __align__(32) unsigned short sVp[4 * 512];

  int id = blockIdx.x;
  int ttile = id & 7; id >>= 3;
  int h = id & 7;     int b = id >> 3;
  int t0 = ttile * 16;
  int lane = threadIdx.x;
  int mtQ = (b * PT + t0) >> 4;

  v16bf qf0 = load_frag(qP + ((size_t)mtQ * 16 + h * 2 + 0) * 512);
  v16bf qf1 = load_frag(qP + ((size_t)mtQ * 16 + h * 2 + 1) * 512);

  const float scale = 0.125f;  // 1/sqrt(64)
  int n = lane & 15;
  int mbase = (lane >> 4) << 3;

  for (int nt = 0; nt < 32; ++nt) {
    stage_t16_async(sK, kvl + (size_t)(b * PN + nt * 16) * (3 * PH) + h * PKEY,
                    3 * PH, 64, lane);
    __syncthreads();
    v8f s = v8f_zero();
    s = wmma_bf16(qf0, load_frag(sK), s);
    s = wmma_bf16(qf1, load_frag(sK + 512), s);
#pragma unroll
    for (int i = 0; i < 8; ++i) {
      int t = t0 + mbase + i;
      int ncol = nt * 16 + n;
      float v = s[i] * scale;
      if (mask[((size_t)b * PT + t) * PN + ncol]) v = -1e9f;
      sScore[(mbase + i) * 512 + ncol] = v;
    }
    __syncthreads();
  }

  // full-wave softmax: lanes r and r+16 split row r's 512 cols, shfl-combine
  {
    int r = lane & 15;
    int c0 = (lane >> 4) << 8;  // 0 or 256
    float m = -3.4e38f;
    for (int c = c0; c < c0 + 256; ++c) m = fmaxf(m, sScore[r * 512 + c]);
    m = fmaxf(m, __shfl_xor(m, 16, 32));
    float ssum = 0.0f;
    for (int c = c0; c < c0 + 256; ++c) {
      float e = __expf(sScore[r * 512 + c] - m);
      ssum += e;
      sScore[r * 512 + c] = e;
    }
    ssum += __shfl_xor(ssum, 16, 32);
    float inv = 1.0f / ssum;
    for (int c = c0; c < c0 + 256; ++c)
      sProbP[(c >> 5) * 512 + frag_idx(r, c & 31)] =
          f2bf(sScore[r * 512 + c] * inv);
  }
  __syncthreads();

  v8f o[4];
#pragma unroll
  for (int j = 0; j < 4; ++j) o[j] = v8f_zero();
  for (int ks = 0; ks < 16; ++ks) {
    for (int i = lane; i < 256; i += 32) {
      int node = i >> 3;
      int f0 = (i & 7) << 3;
      uint4 v = *(const uint4*)(kvl +
          (size_t)(b * PN + ks * 32 + node) * (3 * PH) + PH + h * PKEY + f0);
      const unsigned short* pv = (const unsigned short*)&v;
      int g = (node >> 3) & 1;
      int pos = (node & 7) + 8 * ((node >> 4) & 1);
#pragma unroll
      for (int e = 0; e < 8; ++e) {
        int feat = f0 + e;
        sVp[(feat >> 4) * 512 + ((feat & 15) + 16 * g) * 16 + pos] = pv[e];
      }
    }
    __syncthreads();
    v16bf pf = load_frag(sProbP + ks * 512);
#pragma unroll
    for (int j = 0; j < 4; ++j)
      o[j] = wmma_bf16(pf, load_frag(sVp + j * 512), o[j]);
    __syncthreads();
  }

#pragma unroll
  for (int j = 0; j < 4; ++j)
#pragma unroll
    for (int i = 0; i < 8; ++i) {
      int m = mbase + i;
      int kcol = h * PKEY + j * 16 + n;
      headsP[((size_t)mtQ * 16 + (kcol >> 5)) * 512 + frag_idx(m, kcol & 31)] =
          f2bf(o[j][i]);
    }
}

// ---------------- fused logits + log_softmax ----------------
// One wave per (b, 16-row T tile). grid = B*(T/16) = 2048.
__global__ __launch_bounds__(32)
void logits_kernel(const unsigned short* __restrict__ gP,
                   const unsigned short* __restrict__ kvl,
                   const unsigned char* __restrict__ mask,
                   float* __restrict__ out) {
  __shared__ __align__(32) float          sScore[16 * 512];  // 32 KB
  __shared__ __align__(32) unsigned short sB[16 * 512];      // 16 KB

  int id = blockIdx.x;
  int ttile = id & 7; int b = id >> 3;
  int t0 = ttile * 16;
  int lane = threadIdx.x;
  int mtG = (b * PT + t0) >> 4;
  const float scale = 0.04419417382f;  // 1/sqrt(512)
  int n = lane & 15;
  int mbase = (lane >> 4) << 3;

  // hoist the 16 glimpse A-fragments for this row block (1 wave -> ample VGPRs)
  v16bf afr[16];
#pragma unroll
  for (int kk = 0; kk < 16; ++kk)
    afr[kk] = load_frag(gP + ((size_t)mtG * 16 + kk) * 512);

  for (int nt = 0; nt < 32; ++nt) {
    stage_t16_async(sB, kvl + (size_t)(b * PN + nt * 16) * (3 * PH) + 2 * PH,
                    3 * PH, 512, lane);
    __syncthreads();
    v8f s = v8f_zero();
#pragma unroll
    for (int kk = 0; kk < 16; ++kk)
      s = wmma_bf16(afr[kk], load_frag(sB + kk * 512), s);
#pragma unroll
    for (int i = 0; i < 8; ++i) {
      int t = t0 + mbase + i;
      int ncol = nt * 16 + n;
      float v = fast_tanh(s[i] * scale) * 10.0f;
      if (mask[((size_t)b * PT + t) * PN + ncol]) v = -1e9f;
      sScore[(mbase + i) * 512 + ncol] = v;
    }
    __syncthreads();
  }

  // full-wave log-softmax: lanes r and r+16 split row r, shfl-combine
  {
    int r = lane & 15, t = t0 + r;
    int c0 = (lane >> 4) << 8;
    float m = -3.4e38f;
    for (int c = c0; c < c0 + 256; ++c) m = fmaxf(m, sScore[r * 512 + c]);
    m = fmaxf(m, __shfl_xor(m, 16, 32));
    float ssum = 0.0f;
    for (int c = c0; c < c0 + 256; ++c) ssum += __expf(sScore[r * 512 + c] - m);
    ssum += __shfl_xor(ssum, 16, 32);
    float lse = m + __logf(ssum);
    for (int c = c0; c < c0 + 256; ++c)
      out[((size_t)b * PT + t) * PN + c] = sScore[r * 512 + c] - lse;
  }
}

// ---------------- host launcher ----------------
extern "C" void kernel_launch(void* const* d_in, const int* in_sizes, int n_in,
                              void* d_out, int out_size, void* d_ws, size_t ws_size,
                              hipStream_t stream) {
  const float* emb   = (const float*)d_in[0];
  const int*   cur   = (const int*)d_in[1];
  const float* ucap  = (const float*)d_in[2];
  const float* ubat  = (const float*)d_in[3];
  const float* ctime = (const float*)d_in[4];
  const unsigned char* mask = (const unsigned char*)d_in[5];
  const float* Wc    = (const float*)d_in[6];
  const float* Wkvl  = (const float*)d_in[7];
  const float* Wstep = (const float*)d_in[8];
  const float* Wout  = (const float*)d_in[9];
  float* out = (float*)d_out;

  char* p = (char*)d_ws;
  auto alloc = [&](size_t bytes) -> void* {
    void* r = (void*)p;
    p += (bytes + 255) & ~(size_t)255;
    return r;
  };
  unsigned short* embP   = (unsigned short*)alloc((size_t)PB * PN * PD * 2);
  unsigned short* kvlB   = (unsigned short*)alloc((size_t)PB * PN * 3 * PH * 2);
  unsigned short* WkvlP  = (unsigned short*)alloc((size_t)PD * 3 * PH * 2);
  unsigned short* WstepP = (unsigned short*)alloc((size_t)K2PAD * PH * 2);
  unsigned short* WoutP  = (unsigned short*)alloc((size_t)PH * PH * 2);
  float*          meanE  = (float*)alloc((size_t)PB * PD * 4);
  float*          fixedC = (float*)alloc((size_t)PB * PH * 4);
  unsigned short* nnP    = (unsigned short*)alloc((size_t)PB * PT * K2PAD * 2);
  float*          qF32   = (float*)alloc((size_t)PB * PT * PH * 4);
  unsigned short* qP     = (unsigned short*)alloc((size_t)PB * PT * PH * 2);
  unsigned short* headsP = (unsigned short*)alloc((size_t)PB * PT * PH * 2);
  float*          glimF  = (float*)alloc((size_t)PB * PT * PH * 4);
  unsigned short* glimP  = (unsigned short*)alloc((size_t)PB * PT * PH * 2);

  // 1) packing / prep
  pack_a_kernel<<<8192, 256, 0, stream>>>(emb, embP, PB * PN, PD);
  pack_b_kernel<<<384, 256, 0, stream>>>(Wkvl, WkvlP, PD, 3 * PH, PD);
  pack_b_kernel<<<136, 256, 0, stream>>>(Wstep, WstepP, K2PAD, PH, PD + 3);
  pack_b_kernel<<<128, 256, 0, stream>>>(Wout, WoutP, PH, PH, PH);
  next_nodes_kernel<<<4096, 256, 0, stream>>>(emb, cur, ucap, ubat, ctime, nnP);

  // 2) fixed context
  mean_embed_kernel<<<PB, 256, 0, stream>>>(emb, meanE);
  fixed_context_kernel<<<PB, 256, 0, stream>>>(meanE, Wc, fixedC);

  // 3) kvl = emb @ W_kvlogit  (M=131072, N=1536, K=512) -> row-major bf16
  {
    dim3 grid(3 * PH / 128, PB * PN / 128);
    gemm_packed_kernel<16, 1536, 1><<<grid, 256, 0, stream>>>(embP, WkvlP, kvlB);
  }
  // 4) step projection (M=32768, N=512, K=544) -> f32, then +fixed -> packed bf16
  {
    dim3 grid(PH / 128, PB * PT / 128);
    gemm_packed_kernel<17, 512, 0><<<grid, 256, 0, stream>>>(nnP, WstepP, qF32);
  }
  query_epi_kernel<<<4096, 256, 0, stream>>>(qF32, fixedC, qP);

  // 5) fused multi-head glimpse attention
  attention_kernel<<<PB * PHEADS * (PT / 16), 32, 0, stream>>>(qP, kvlB, mask,
                                                               headsP);

  // 6) glimpse = heads_out @ W_out (M=32768, N=512, K=512) -> f32 -> packed bf16
  {
    dim3 grid(PH / 128, PB * PT / 128);
    gemm_packed_kernel<16, 512, 0><<<grid, 256, 0, stream>>>(headsP, WoutP, glimF);
  }
  pack_a_kernel<<<4096, 256, 0, stream>>>(glimF, glimP, PB * PT, PH);

  // 7) logits + tanh clip + mask + log_softmax
  logits_kernel<<<PB * (PT / 16), 32, 0, stream>>>(glimP, kvlB, mask, out);
}
